// SwinTransformerBlock_62130996904238
// MI455X (gfx1250) — matrix operand order repaired and checked by reading, not compiled, not run on hardware
//
#include <hip/hip_runtime.h>

// ---------------------------------------------------------------------------
// MI455X (gfx1250) implementation. All matmuls via v_wmma_f32_16x16x32_bf16.
// Activations live in pixel-major [N][C] fp32; weights converted once to a
// bf16 [n][k] arena. Shifted-window attention folds roll/window/unroll into
// gather/scatter index math (pointwise ops commute with jnp.roll).
// GEMM: A tile register-double-buffered + packed to LDS bf16; B tile staged
// with GLOBAL_LOAD_ASYNC_TO_LDS_B128 (ASYNCcnt) to overlap with the WMMA pipe.
// ---------------------------------------------------------------------------

typedef __attribute__((ext_vector_type(16))) __bf16 v16bf;
typedef __attribute__((ext_vector_type(8)))  float  v8f;

union Frag { uint4 u[2]; v16bf v; };

#define NPIX 65536   // B*H*W
#define HIMG 128
#define WIMG 128
#define DIMC 320

__device__ __forceinline__ unsigned short f2bf(float f) {
  union { float f; unsigned u; } w; w.f = f;
  unsigned u = w.u + 0x7FFFu + ((w.u >> 16) & 1u);   // round-to-nearest-even
  return (unsigned short)(u >> 16);
}
__device__ __forceinline__ float gelu_f(float x) {
  return 0.5f * x * (1.f + erff(x * 0.70710678118654752f));
}
// LDS byte offset of a shared-memory pointer (generic ptr low 32 bits, ISA 10.2)
__device__ __forceinline__ unsigned lds_off(const void* p) {
  return (unsigned)(size_t)p;
}

// ------------------------------ weight fp32 -> bf16 ------------------------
__global__ void wconv_kernel(const float* __restrict__ src, unsigned short* __restrict__ dst,
                             int rows, int cols, int trans) {
  int idx = blockIdx.x * blockDim.x + threadIdx.x;
  if (idx >= rows * cols) return;
  int r = idx / cols, c = idx % cols;
  unsigned short v = f2bf(src[idx]);
  if (trans) dst[(size_t)c * rows + r] = v;   // io -> [out][in]
  else       dst[idx] = v;                    // oc -> [out][in] already
}

// ------------------------------ layout transposes --------------------------
__global__ __launch_bounds__(256) void nchw2nhwc(const float* __restrict__ in,
                                                 float* __restrict__ out) {
  __shared__ float tile[32][33];
  int hw0 = blockIdx.x * 32, c0 = blockIdx.y * 32, b = blockIdx.z;
  int tx = threadIdx.x & 31, ty = threadIdx.x >> 5;
  for (int i = 0; i < 4; ++i) {
    int c = c0 + ty + i * 8;
    tile[ty + i * 8][tx] = in[((size_t)(b * DIMC + c) << 14) + hw0 + tx];
  }
  __syncthreads();
  for (int i = 0; i < 4; ++i) {
    int hw = hw0 + ty + i * 8;
    out[(size_t)((b << 14) + hw) * DIMC + c0 + tx] = tile[tx][ty + i * 8];
  }
}
__global__ __launch_bounds__(256) void nhwc2nchw(const float* __restrict__ in,
                                                 float* __restrict__ out) {
  __shared__ float tile[32][33];
  int hw0 = blockIdx.x * 32, c0 = blockIdx.y * 32, b = blockIdx.z;
  int tx = threadIdx.x & 31, ty = threadIdx.x >> 5;
  for (int i = 0; i < 4; ++i) {
    int hw = hw0 + ty + i * 8;
    tile[ty + i * 8][tx] = in[(size_t)((b << 14) + hw) * DIMC + c0 + tx];  // [hwl][cl]
  }
  __syncthreads();
  for (int i = 0; i < 4; ++i) {
    int c = c0 + ty + i * 8;
    out[((size_t)(b * DIMC + c) << 14) + hw0 + tx] = tile[tx][ty + i * 8];
  }
}

// ------------------------------ layernorm (wave per pixel) -----------------
__global__ __launch_bounds__(256) void ln_kernel(const float* __restrict__ in,
    const float* __restrict__ g, const float* __restrict__ b,
    float* __restrict__ out, int C) {
  int wv = threadIdx.x >> 5, ln = threadIdx.x & 31;
  size_t p = (size_t)blockIdx.x * 8 + wv;
  const float* row = in + p * C;
  float s = 0.f;
  for (int c = ln; c < C; c += 32) s += row[c];
  for (int o = 16; o > 0; o >>= 1) s += __shfl_xor(s, o, 32);
  float mu = s / C;
  float var = 0.f;
  for (int c = ln; c < C; c += 32) { float d = row[c] - mu; var += d * d; }
  for (int o = 16; o > 0; o >>= 1) var += __shfl_xor(var, o, 32);
  float rs = rsqrtf(var / C + 1e-5f);
  float* orow = out + p * C;
  for (int c = ln; c < C; c += 32) orow[c] = (row[c] - mu) * rs * g[c] + b[c];
}

// ------------------------------ bf16 WMMA GEMM -----------------------------
// Co[M][N] = act(A[M][K]) @ Bw[N][K]^T + bias (+addC), opt. output activation.
// Block tile 128x64, 8 waves (4x2), 32x32 per wave, K stepped by 32.
// A: fp32 global -> regs (double buffered) -> bf16 LDS (pitch 40).
// B: bf16 global -> LDS via global_load_async_to_lds_b128 (ASYNCcnt).
#define GEMM_EPI(ACC, MI, NJ) do {                                          \
    int ng = col0 + wn * 32 + (NJ) * 16 + l16;                              \
    float bvv = bias ? bias[ng] : 0.f;                                      \
    for (int r = 0; r < 8; ++r) {                                           \
      int mg = row0 + wm * 32 + (MI) * 16 + r + hf * 8;                     \
      float vv = ACC[r] + bvv;                                              \
      if (addC) vv += addC[(size_t)mg * ld_add + ng];                       \
      if (out_act == 1) vv = 1.f / (1.f + __expf(-vv));                     \
      else if (out_act == 2) vv = tanhf(vv);                                \
      Co[(size_t)mg * ldc + ng] = vv;                                       \
    }                                                                       \
  } while (0)

__global__ __launch_bounds__(256) void gemm_bf16_kernel(
    const float* __restrict__ A, int lda,
    const unsigned short* __restrict__ Bw,
    const float* __restrict__ bias,
    const float* __restrict__ addC, int ld_add,
    float* __restrict__ Co, int ldc,
    int K, int a_act, int out_act) {
  __shared__ unsigned short As[128 * 40];   // 128 rows x 32 k, pitch 40
  __shared__ unsigned short Bs[64 * 40];    // 64 cols x 32 k, pitch 40
  const int tid = threadIdx.x;
  const int lane = tid & 31, wave = tid >> 5;
  const int wm = wave >> 1, wn = wave & 1;
  const int l16 = lane & 15, hf = lane >> 4;
  const int row0 = blockIdx.x * 128, col0 = blockIdx.y * 64;

  v8f acc00, acc01, acc10, acc11;
#pragma unroll
  for (int i = 0; i < 8; ++i) { acc00[i] = 0.f; acc01[i] = 0.f; acc10[i] = 0.f; acc11[i] = 0.f; }

  const int lrow = tid >> 3;            // A stage: 32 rows x (8 thr * 4 f32)
  const int lk = (tid & 7) << 2;
  const int bn = tid >> 2;              // B stage: 64 cols x (4 thr * 8 bf16)
  const int bk = (tid & 3) << 3;

  float4 pf[4];                          // register prefetch of A tile
#pragma unroll
  for (int i = 0; i < 4; ++i)
    pf[i] = *(const float4*)(A + (size_t)(row0 + lrow + i * 32) * lda + lk);

  for (int k0 = 0; k0 < K; k0 += 32) {
    __syncthreads();                     // protect previous-iteration LDS reads
    // ---- async-stage B tile (one instruction per wave, ASYNCcnt tracked) ----
    {
      const unsigned short* gp = Bw + (size_t)(col0 + bn) * K + k0 + bk;
      unsigned lo = lds_off(&Bs[bn * 40 + bk]);
      asm volatile("global_load_async_to_lds_b128 %0, %1, off"
                   :: "v"(lo), "v"(gp) : "memory");
    }
    // ---- pack prefetched A tile to bf16 LDS ----
#pragma unroll
    for (int i = 0; i < 4; ++i) {
      float4 f = pf[i];
      if (a_act == 3) { f.x = gelu_f(f.x); f.y = gelu_f(f.y); f.z = gelu_f(f.z); f.w = gelu_f(f.w); }
      uint2 pk;
      pk.x = (unsigned)f2bf(f.x) | ((unsigned)f2bf(f.y) << 16);
      pk.y = (unsigned)f2bf(f.z) | ((unsigned)f2bf(f.w) << 16);
      *(uint2*)&As[(lrow + i * 32) * 40 + lk] = pk;
    }
    // ---- prefetch next A tile while WMMAs run ----
    if (k0 + 32 < K) {
#pragma unroll
      for (int i = 0; i < 4; ++i)
        pf[i] = *(const float4*)(A + (size_t)(row0 + lrow + i * 32) * lda + (k0 + 32 + lk));
    }
    asm volatile("s_wait_asynccnt 0x0" ::: "memory");
    __syncthreads();

    Frag a0, a1, b0, b1;
    const unsigned short* ap0 = &As[(wm * 32 + l16) * 40 + hf * 8];
    a0.u[0] = *(const uint4*)ap0;  a0.u[1] = *(const uint4*)(ap0 + 16);
    const unsigned short* ap1 = &As[(wm * 32 + 16 + l16) * 40 + hf * 8];
    a1.u[0] = *(const uint4*)ap1;  a1.u[1] = *(const uint4*)(ap1 + 16);
    const unsigned short* bp0 = &Bs[(wn * 32 + l16) * 40 + hf * 8];
    b0.u[0] = *(const uint4*)bp0;  b0.u[1] = *(const uint4*)(bp0 + 16);
    const unsigned short* bp1 = &Bs[(wn * 32 + 16 + l16) * 40 + hf * 8];
    b1.u[0] = *(const uint4*)bp1;  b1.u[1] = *(const uint4*)(bp1 + 16);

    acc00 = __builtin_amdgcn_wmma_f32_16x16x32_bf16(false, a0.v, false, b0.v, (short)0, acc00, false, false);
    acc01 = __builtin_amdgcn_wmma_f32_16x16x32_bf16(false, a0.v, false, b1.v, (short)0, acc01, false, false);
    acc10 = __builtin_amdgcn_wmma_f32_16x16x32_bf16(false, a1.v, false, b0.v, (short)0, acc10, false, false);
    acc11 = __builtin_amdgcn_wmma_f32_16x16x32_bf16(false, a1.v, false, b1.v, (short)0, acc11, false, false);
  }
  GEMM_EPI(acc00, 0, 0); GEMM_EPI(acc01, 0, 1);
  GEMM_EPI(acc10, 1, 0); GEMM_EPI(acc11, 1, 1);
}

// ------------------------------ depthwise conv -----------------------------
__global__ void dwconv_kernel(const float* __restrict__ in, int ldi, int icoff,
                              const float* __restrict__ w, const float* __restrict__ b,
                              float* __restrict__ out, int ldo, int ocoff,
                              int C, int KH, int KW, int PH, int PW, int total) {
  int idx = blockIdx.x * blockDim.x + threadIdx.x;
  if (idx >= total) return;
  int c = idx % C, p = idx / C;
  int bi = p >> 14, hw = p & 16383, h = hw >> 7, ww = hw & 127;
  float acc = b[c];
  for (int kh = 0; kh < KH; ++kh) {
    int hh = h + kh - PH;
    if (hh < 0 || hh >= HIMG) continue;
    for (int kw = 0; kw < KW; ++kw) {
      int wq = ww + kw - PW;
      if (wq < 0 || wq >= WIMG) continue;
      acc += in[(size_t)((bi << 14) + (hh << 7) + wq) * ldi + icoff + c] * w[c * KH * KW + kh * KW + kw];
    }
  }
  out[(size_t)p * ldo + ocoff + c] = acc;
}

// ------------------------------ windowed attention -------------------------
// One block per window; T/16 waves; S=QK^T and O=attn*V via WMMA.
template <int T, int C>
__global__ __launch_bounds__(128) void attn_kernel(
    const float* __restrict__ xin, const float* __restrict__ qf,
    const float* __restrict__ vf,  const float* __restrict__ lepe,
    const float* __restrict__ attn_in, const float* __restrict__ temp,
    float* __restrict__ a_out,
    float* __restrict__ outp, int ldo, int ocoff,
    int Hs, int Ws, float scale) {
  constexpr int TP = (T < 32) ? 32 : T;
  constexpr int NW = T / 16;
  constexpr int NC = C / 16;
  __shared__ unsigned short qs[T * C];
  __shared__ unsigned short ks[T * C];
  __shared__ unsigned short vs[C * TP];   // vT, zero-padded K
  __shared__ unsigned short as2[T * TP];  // attn bf16, zero-padded K
  __shared__ int pidx[T];

  const int wi = blockIdx.x, tid = threadIdx.x, nthr = blockDim.x;
  const int nbx = WIMG / Ws, nby = HIMG / Hs;
  const int b = wi / (nby * nbx);
  const int rem = wi - b * (nby * nbx);
  const int hy = rem / nbx, wx = rem % nbx;

  if (tid < T) {   // shifted-window gather index (roll folded in)
    int tlh = tid / Ws, tlw = tid % Ws;
    int ho = (hy * Hs + tlh + (Hs >> 1)) & (HIMG - 1);
    int wo = (wx * Ws + tlw + (Ws >> 1)) & (WIMG - 1);
    pidx[tid] = (b << 14) + (ho << 7) + wo;
  }
  __syncthreads();

  for (int idx = tid; idx < T * C; idx += nthr) {
    int t = idx / C, c = idx - t * C;
    size_t pp = (size_t)pidx[t] * C + c;
    float lp = lepe[pp];
    qs[idx] = f2bf((qf[pp] + lp) * scale);
    ks[idx] = f2bf(xin[pp] + lp);
  }
  for (int idx = tid; idx < C * TP; idx += nthr) {
    int c = idx / TP, t = idx - c * TP;
    vs[idx] = f2bf((t < T) ? vf[(size_t)pidx[t] * C + c] : 0.f);
  }
  __syncthreads();

  const int lane = tid & 31, w = tid >> 5;
  const int l16 = lane & 15, hf = lane >> 4;

  // ---- S = qs @ ks^T  (T x T) ----
  v8f sacc[NW];
#pragma unroll
  for (int j = 0; j < NW; ++j)
#pragma unroll
    for (int i = 0; i < 8; ++i) sacc[j][i] = 0.f;
#pragma unroll
  for (int k0 = 0; k0 < C; k0 += 32) {
    Frag af;
    const unsigned short* ap = &qs[(w * 16 + l16) * C + k0 + hf * 8];
    af.u[0] = *(const uint4*)ap;  af.u[1] = *(const uint4*)(ap + 16);
#pragma unroll
    for (int j = 0; j < NW; ++j) {
      Frag bf;
      const unsigned short* bp = &ks[(j * 16 + l16) * C + k0 + hf * 8];
      bf.u[0] = *(const uint4*)bp;  bf.u[1] = *(const uint4*)(bp + 16);
      sacc[j] = __builtin_amdgcn_wmma_f32_16x16x32_bf16(false, af.v, false, bf.v, (short)0, sacc[j], false, false);
    }
  }

  // ---- softmax rows + temp*attn_in (row spread over 16 lanes in C-layout) ----
  float tmpv = temp[0];
#pragma unroll
  for (int r = 0; r < 8; ++r) {
    int rowg = w * 16 + r + 8 * hf;
    float mx = -1e30f;
#pragma unroll
    for (int j = 0; j < NW; ++j) mx = fmaxf(mx, sacc[j][r]);
    for (int o = 8; o > 0; o >>= 1) mx = fmaxf(mx, __shfl_xor(mx, o, 16));
    float ex[NW], sum = 0.f;
#pragma unroll
    for (int j = 0; j < NW; ++j) { ex[j] = __expf(sacc[j][r] - mx); sum += ex[j]; }
    for (int o = 8; o > 0; o >>= 1) sum += __shfl_xor(sum, o, 16);
    float inv = 1.f / sum;
#pragma unroll
    for (int j = 0; j < NW; ++j) {
      int colg = j * 16 + l16;
      size_t aoff = (size_t)wi * (T * T) + (size_t)rowg * T + colg;
      float av = ex[j] * inv + tmpv * attn_in[aoff];
      a_out[aoff] = av;
      as2[rowg * TP + colg] = f2bf(av);
      if (T < TP) as2[rowg * TP + 16 + colg] = 0;   // zero-pad K (T==16 only)
    }
  }
  __syncthreads();

  // ---- O = attn @ V  (T x C), K = TP ----
  v8f oacc[NC];
#pragma unroll
  for (int j = 0; j < NC; ++j)
#pragma unroll
    for (int i = 0; i < 8; ++i) oacc[j][i] = 0.f;
#pragma unroll
  for (int k0 = 0; k0 < TP; k0 += 32) {
    Frag af;
    const unsigned short* ap = &as2[(w * 16 + l16) * TP + k0 + hf * 8];
    af.u[0] = *(const uint4*)ap;  af.u[1] = *(const uint4*)(ap + 16);
#pragma unroll
    for (int j = 0; j < NC; ++j) {
      Frag bf;
      const unsigned short* bp = &vs[(j * 16 + l16) * TP + k0 + hf * 8];
      bf.u[0] = *(const uint4*)bp;  bf.u[1] = *(const uint4*)(bp + 16);
      oacc[j] = __builtin_amdgcn_wmma_f32_16x16x32_bf16(false, af.v, false, bf.v, (short)0, oacc[j], false, false);
    }
  }
#pragma unroll
  for (int j = 0; j < NC; ++j) {
    int c = j * 16 + l16;
#pragma unroll
    for (int r = 0; r < 8; ++r) {
      int rowg = w * 16 + r + 8 * hf;
      int pp = pidx[rowg];   // scatter back to original pixel (+ lepe)
      outp[(size_t)pp * ldo + ocoff + c] = oacc[j][r] + lepe[(size_t)pp * C + c];
    }
  }
}

// ------------------------------ gate elementwise ---------------------------
__global__ void gate_prep_kernel(const float* __restrict__ s, int lds_, int scoff,
                                 const float* __restrict__ addp, int lda_, int acoff,
                                 const float* __restrict__ hid, int ldh, int hcoff,
                                 float* __restrict__ g) {
  int idx = blockIdx.x * blockDim.x + threadIdx.x;
  if (idx >= NPIX * 192) return;
  int p = idx / 192, c = idx - p * 192;
  float v;
  if (c < 64) {
    v = s[(size_t)p * lds_ + scoff + c];
    if (addp) v += addp[(size_t)p * lda_ + acoff + c];
  } else {
    v = hid[(size_t)p * ldh + hcoff + (c - 64)];
  }
  g[(size_t)p * 192 + c] = v;
}
__global__ void gate_mulr_kernel(float* __restrict__ g, const float* __restrict__ r) {
  int idx = blockIdx.x * blockDim.x + threadIdx.x;
  if (idx >= NPIX * 128) return;
  int p = idx >> 7, c = idx & 127;
  g[(size_t)p * 192 + 64 + c] *= r[(size_t)p * 128 + c];
}
__global__ void gate_combine_kernel(const float* __restrict__ z, const float* __restrict__ e,
                                    const float* __restrict__ hh, float* __restrict__ go) {
  int idx = blockIdx.x * blockDim.x + threadIdx.x;
  if (idx >= NPIX * 64) return;
  float zz = z[idx];
  go[idx] = (1.f - zz) * e[idx] + zz * hh[idx];
}

// ===========================================================================
extern "C" void kernel_launch(void* const* d_in, const int* in_sizes, int n_in,
                              void* d_out, int out_size, void* d_ws, size_t ws_size,
                              hipStream_t stream) {
  (void)in_sizes; (void)n_in; (void)out_size; (void)ws_size;
  // Input order: x, attn1_in..attn4_in, then params flattened in setup_inputs()
  // insertion order (depth-first through the nested dicts).
  auto P = [&](int i) { return (const float*)d_in[i]; };
  const float* x_nchw = P(0);
  float* out_f = (float*)d_out;
  const size_t OFF_A1 = 20971520, OFF_A2 = 25165824, OFF_A3 = 27262976, OFF_A4 = 29360128;

  // ---- bf16 weight arena at start of d_ws ----
  unsigned short* warena = (unsigned short*)d_ws;
  size_t woff = 0;
  auto walloc = [&](size_t n) { size_t o = woff; woff += (n + 15) & ~(size_t)15; return o; };
  size_t oWrt = walloc(128 * 320), oWp1 = walloc(192 * 320), oWpr = walloc(320 * 320);
  size_t oWq1 = walloc(128 * 128), oWv1 = walloc(128 * 128);
  size_t oWq2 = walloc(64 * 64), oWv2 = walloc(64 * 64);
  size_t oWq3 = walloc(64 * 64), oWv3 = walloc(64 * 64);
  size_t oWq4 = walloc(64 * 64), oWv4 = walloc(64 * 64);
  size_t oWr[3], oWz[3], oWf[3], oWe[3];
  for (int g = 0; g < 3; ++g) {
    oWr[g] = walloc(128 * 192); oWz[g] = walloc(64 * 192);
    oWf[g] = walloc(64 * 192);  oWe[g] = walloc(64 * 128);
  }
  size_t oF1 = walloc(640 * 320), oF2 = walloc(320 * 640);

  // ---- fp32 activation arena (regions reused across the serialized pipeline)
  float* fb = (float*)((char*)d_ws + ((woff * 2 + 255) & ~(size_t)255));
  const size_t NP = NPIX;
  float* xnhwc = fb;                    // 320N  (later: first half of h)
  float* xcat  = xnhwc + 320 * NP;      // 320N  x1|x2|x3|x4 (later: 2nd half of h)
  float* xn    = xcat + 320 * NP;       // 320N  xn -> y -> final (in-place fc2)
  float* xw    = xn + 320 * NP;         // 192N  s0|s1|s2   (later: yn, with xr)
  float* xr    = xw + 192 * NP;         // 128N  router/attn1 input
  float* bq    = xr + 128 * NP;         // 128N  q           (later: h2 region)
  float* bv    = bq + 128 * NP;         // 128N  v
  float* ble   = bv + 128 * NP;         // 128N  lepe
  float* bg    = ble + 128 * NP;        // 192N  gate concat buffer
  float* brr   = bg + 192 * NP;         // 128N  r
  float* bz    = brr + 128 * NP;        // 64N
  float* bhh   = bz + 64 * NP;          // 64N
  float* bee   = bhh + 64 * NP;         // 64N
  float* bgo   = bee + 64 * NP;         // 64N  gate output (attn input)
  float* bdt   = bgo + 64 * NP;         // 160N dw4 temp
  float* yn    = xw;                    // 320N (xw+xr region)
  float* hbuf  = xnhwc;                 // 640N (xnhwc+xcat region)
  float* h2    = bq;                    // 640N (bq..brr region, 704N available)

  auto GEMM = [&](const float* A, int lda, size_t bo, const float* bias,
                  const float* addC, int ldadd, float* Cp, int ldc,
                  int Nn, int Kk, int a_act, int o_act) {
    dim3 grid(NPIX / 128, Nn / 64);
    gemm_bf16_kernel<<<grid, 256, 0, stream>>>(A, lda, warena + bo, bias,
                                               addC, ldadd, Cp, ldc, Kk, a_act, o_act);
  };
  auto WC = [&](int pi, size_t off, int rows, int cols, int trans) {
    int tot = rows * cols;
    wconv_kernel<<<(tot + 255) / 256, 256, 0, stream>>>(P(pi), warena + off, rows, cols, trans);
  };
  auto DW = [&](const float* in, int ldi, int icoff, int wpi, int bpi,
                float* out, int ldo, int ocoff, int C, int KH, int KW, int PH, int PW) {
    int tot = NPIX * C;
    dwconv_kernel<<<(tot + 255) / 256, 256, 0, stream>>>(in, ldi, icoff, P(wpi), P(bpi),
                                                         out, ldo, ocoff, C, KH, KW, PH, PW, tot);
  };

  // ---- 0) weight conversion (oc weights direct; io weights transposed) ----
  WC(9, oWrt, 128, 320, 0);  WC(11, oWp1, 192, 320, 0);  WC(13, oWpr, 320, 320, 1);
  WC(15, oWq1, 128, 128, 0); WC(17, oWv1, 128, 128, 0);
  WC(22, oWq2, 64, 64, 0);   WC(24, oWv2, 64, 64, 0);
  WC(29, oWq3, 64, 64, 0);   WC(31, oWv3, 64, 64, 0);
  WC(36, oWq4, 64, 64, 0);   WC(38, oWv4, 64, 64, 0);
  for (int g = 0; g < 3; ++g) {
    int base = 43 + g * 8;
    WC(base + 0, oWr[g], 128, 192, 0); WC(base + 2, oWz[g], 64, 192, 0);
    WC(base + 4, oWf[g], 64, 192, 0);  WC(base + 6, oWe[g], 64, 128, 0);
  }
  WC(67, oF1, 320, 640, 1);  WC(69, oF2, 640, 320, 1);

  // ---- 1) NCHW->NHWC, LN1, router/window projections ----
  nchw2nhwc<<<dim3(512, 10, 4), 256, 0, stream>>>(x_nchw, xnhwc);
  ln_kernel<<<NPIX / 8, 256, 0, stream>>>(xnhwc, P(5), P(6), xn, 320);
  GEMM(xn, 320, oWrt, P(10), nullptr, 0, xr, 128, 128, 320, 0, 0);
  GEMM(xn, 320, oWp1, P(12), nullptr, 0, xw, 192, 192, 320, 0, 0);

  // ---- 2) attn1: 8x8 windows, C=128 ----
  GEMM(xr, 128, oWq1, P(16), nullptr, 0, bq, 128, 128, 128, 0, 0);
  GEMM(xr, 128, oWv1, P(18), nullptr, 0, bv, 128, 128, 128, 0, 0);
  DW(bv, 128, 0, 19, 20, ble, 128, 0, 128, 5, 5, 2, 2);
  attn_kernel<64, 128><<<1024, 128, 0, stream>>>(xr, bq, bv, ble, P(1), P(21),
      out_f + OFF_A1, xcat, 320, 0, 8, 8, 0.08838834764831845f);

  // ---- 3) gate1 -> attn2 (8x4, C=64) ----
  gate_prep_kernel<<<(NPIX * 192 + 255) / 256, 256, 0, stream>>>(xw, 192, 0, nullptr, 0, 0, xcat, 320, 0, bg);
  GEMM(bg, 192, oWr[0], P(44), nullptr, 0, brr, 128, 128, 192, 0, 1);
  GEMM(bg, 192, oWz[0], P(46), nullptr, 0, bz, 64, 64, 192, 0, 1);
  gate_mulr_kernel<<<(NPIX * 128 + 255) / 256, 256, 0, stream>>>(bg, brr);
  GEMM(bg, 192, oWf[0], P(48), nullptr, 0, bhh, 64, 64, 192, 0, 2);
  GEMM(xcat, 320, oWe[0], P(50), nullptr, 0, bee, 64, 64, 128, 0, 0);
  gate_combine_kernel<<<(NPIX * 64 + 255) / 256, 256, 0, stream>>>(bz, bee, bhh, bgo);

  GEMM(bgo, 64, oWq2, P(23), nullptr, 0, bq, 64, 64, 64, 0, 0);
  GEMM(bgo, 64, oWv2, P(25), nullptr, 0, bv, 64, 64, 64, 0, 0);
  DW(bv, 64, 0, 26, 27, ble, 64, 0, 64, 5, 5, 2, 2);
  attn_kernel<32, 64><<<2048, 64, 0, stream>>>(bgo, bq, bv, ble, P(2), P(28),
      out_f + OFF_A2, xcat, 320, 128, 8, 4, 0.125f);

  // ---- 4) gate2 -> attn3 (4x8, C=64) ----
  gate_prep_kernel<<<(NPIX * 192 + 255) / 256, 256, 0, stream>>>(xw, 192, 64, xcat, 320, 128, xcat, 320, 0, bg);
  GEMM(bg, 192, oWr[1], P(52), nullptr, 0, brr, 128, 128, 192, 0, 1);
  GEMM(bg, 192, oWz[1], P(54), nullptr, 0, bz, 64, 64, 192, 0, 1);
  gate_mulr_kernel<<<(NPIX * 128 + 255) / 256, 256, 0, stream>>>(bg, brr);
  GEMM(bg, 192, oWf[1], P(56), nullptr, 0, bhh, 64, 64, 192, 0, 2);
  GEMM(xcat, 320, oWe[1], P(58), nullptr, 0, bee, 64, 64, 128, 0, 0);
  gate_combine_kernel<<<(NPIX * 64 + 255) / 256, 256, 0, stream>>>(bz, bee, bhh, bgo);

  GEMM(bgo, 64, oWq3, P(30), nullptr, 0, bq, 64, 64, 64, 0, 0);
  GEMM(bgo, 64, oWv3, P(32), nullptr, 0, bv, 64, 64, 64, 0, 0);
  DW(bv, 64, 0, 33, 34, ble, 64, 0, 64, 5, 5, 2, 2);
  attn_kernel<32, 64><<<2048, 64, 0, stream>>>(bgo, bq, bv, ble, P(3), P(35),
      out_f + OFF_A3, xcat, 320, 192, 4, 8, 0.125f);

  // ---- 5) gate3 -> attn4 (4x4, C=64) ----
  gate_prep_kernel<<<(NPIX * 192 + 255) / 256, 256, 0, stream>>>(xw, 192, 128, xcat, 320, 192, xcat, 320, 0, bg);
  GEMM(bg, 192, oWr[2], P(60), nullptr, 0, brr, 128, 128, 192, 0, 1);
  GEMM(bg, 192, oWz[2], P(62), nullptr, 0, bz, 64, 64, 192, 0, 1);
  gate_mulr_kernel<<<(NPIX * 128 + 255) / 256, 256, 0, stream>>>(bg, brr);
  GEMM(bg, 192, oWf[2], P(64), nullptr, 0, bhh, 64, 64, 192, 0, 2);
  GEMM(xcat, 320, oWe[2], P(66), nullptr, 0, bee, 64, 64, 128, 0, 0);
  gate_combine_kernel<<<(NPIX * 64 + 255) / 256, 256, 0, stream>>>(bz, bee, bhh, bgo);

  GEMM(bgo, 64, oWq4, P(37), nullptr, 0, bq, 64, 64, 64, 0, 0);
  GEMM(bgo, 64, oWv4, P(39), nullptr, 0, bv, 64, 64, 64, 0, 0);
  DW(bv, 64, 0, 40, 41, ble, 64, 0, 64, 5, 5, 2, 2);
  attn_kernel<16, 64><<<4096, 32, 0, stream>>>(bgo, bq, bv, ble, P(4), P(42),
      out_f + OFF_A4, xcat, 320, 256, 4, 4, 0.125f);

  // ---- 6) projection + residual (epilogue add), LN2, MLP ----
  GEMM(xcat, 320, oWpr, P(14), xnhwc, 320, xn, 320, 320, 320, 0, 0);   // y
  ln_kernel<<<NPIX / 8, 256, 0, stream>>>(xn, P(7), P(8), yn, 320);
  GEMM(yn, 320, oF1, P(68), nullptr, 0, hbuf, 640, 640, 320, 0, 0);    // fc1
  DW(hbuf, 640,   0, 71, 72, h2, 640,   0, 160, 3, 3, 1, 1);
  DW(hbuf, 640, 160, 73, 74, h2, 640, 160, 160, 5, 3, 2, 1);
  DW(hbuf, 640, 320, 75, 76, h2, 640, 320, 160, 3, 5, 1, 2);
  DW(hbuf, 640, 480, 77, 78, bdt, 160,  0, 160, 3, 3, 1, 1);
  DW(bdt,  160,   0, 79, 80, h2, 640, 480, 160, 3, 3, 1, 1);
  // fc2 with GELU fused on the A operand, residual y added in epilogue, in place
  GEMM(h2, 640, oF2, P(70), xn, 320, xn, 320, 320, 640, 3, 0);

  // ---- 7) final NHWC -> NCHW into d_out ----
  nhwc2nchw<<<dim3(512, 10, 4), 256, 0, stream>>>(xn, out_f);
}
